// ScaledDotAttention_55233279426704
// MI455X (gfx1250) — matrix operand order repaired and checked by reading, not compiled, not run on hardware
//
#include <hip/hip_runtime.h>
#include <hip/hip_bf16.h>

// Problem constants (reference: Q=4096, M=8192, D=128, fp32 in/out)
#define Qn 4096
#define Mn 8192
#define Dn 128
#define BQ 128              // query rows per block (8 waves x 16 rows)
#define BM 64               // memory rows per inner tile
#define SPLITS 8            // flash-decoding splits over M
#define MSPLIT (Mn / SPLITS)    // 1024
#define NT1 (BM / 16)       // 4  n-tiles in GEMM1
#define NKC (Dn / 32)       // 4  k-chunks (K=32) in GEMM1
#define NDT (Dn / 16)       // 8  n-tiles in GEMM2
#define NKC2 (BM / 32)      // 2  k-chunks in GEMM2
#define KPAD 8              // LDS padding (halves) to break bank aliasing
#define VPAD 8

typedef __attribute__((ext_vector_type(16))) _Float16 v16h;
typedef __attribute__((ext_vector_type(8)))  float    v8f;

__device__ __forceinline__ v8f wmma_f16(v16h a, v16h b, v8f c) {
  // D = A(16x32 f16) * B(32x16 f16) + C(16x16 f32)
  return __builtin_amdgcn_wmma_f32_16x16x32_f16(
      /*neg_a=*/false, a, /*neg_b=*/false, b,
      /*c_mod=*/(short)0, c, /*reuse_a=*/false, /*reuse_b=*/false);
}

// Pass 1: per (q-block, m-split) partial flash attention.
// Emits unnormalized O_part plus per-row running max m and sum l.
__global__ __launch_bounds__(256)
void attn_partial(const float* __restrict__ Qm,   // [Qn, Dn]
                  const float* __restrict__ Km,   // [Mn, Dn] (keys AND values)
                  const float* __restrict__ Wm,   // [Qn, Mn]
                  float* __restrict__ Opart,      // [SPLITS, Qn, Dn]
                  float* __restrict__ Mst,        // [SPLITS, Qn]
                  float* __restrict__ Lst)        // [SPLITS, Qn]
{
  // Cooperative f16 staging of the K/V tile, converted once per block:
  //   Kt: row-major   [mem_row][d]  -> GEMM1 B-fragments (contiguous in d)
  //   Vt: transposed  [d][mem_row]  -> GEMM2 B-fragments (contiguous in k)
  __shared__ _Float16 Kt[BM][Dn + KPAD];      // ~17 KB
  __shared__ _Float16 Vt[Dn][BM + VPAD];      // ~18 KB
  __shared__ _Float16 Plds[8][16][BM];        // 16 KB: per-wave P layout swap

  const int tid  = threadIdx.x;
  const int wave = tid >> 5;
  const int lane = tid & 31;
  const int half = lane >> 4;   // which 16-lane half of the wave32
  const int lm   = lane & 15;
  const int q0   = blockIdx.x * BQ + wave * 16;   // this wave's 16-row strip
  const int split = blockIdx.y;
  const int mbeg  = split * MSPLIT;

  const float scale = 0.08838834764831845f;  // 1/sqrt(128)

  // ---- load Q strip once as WMMA A-fragments (f32 -> f16) ----
  // A layout (16x32 f16): lane holds row (lane&15); K = half*16+{0..15}
  v16h qa[NKC];
  {
    const float* qp = Qm + (size_t)(q0 + lm) * Dn;
    #pragma unroll
    for (int kk = 0; kk < NKC; ++kk) {
      #pragma unroll
      for (int j = 0; j < 8; ++j) {
        qa[kk][j]     = (_Float16)qp[kk * 32 + half * 16 + j];
        qa[kk][8 + j] = (_Float16)qp[kk * 32 + half * 16 + 8 + j];
      }
    }
  }

  v8f zero = {};
  v8f o[NDT];
  #pragma unroll
  for (int dt = 0; dt < NDT; ++dt) o[dt] = zero;

  // per-row online-softmax state; C-layout row = r + 8*half
  float mrow[8], lrow[8];
  #pragma unroll
  for (int r = 0; r < 8; ++r) { mrow[r] = -3.0e30f; lrow[r] = 0.0f; }

  for (int m0 = mbeg; m0 < mbeg + MSPLIT; m0 += BM) {
    // ---- cooperative: stage K/V tile into LDS as f16 (one convert/element) ----
    __syncthreads();
    for (int i = tid; i < BM * Dn; i += 256) {
      int k = i >> 7;          // Dn == 128
      int d = i & (Dn - 1);
      _Float16 h = (_Float16)Km[(size_t)(m0 + k) * Dn + d];  // coalesced (L2 hit)
      Kt[k][d] = h;
      Vt[d][k] = h;
    }
    __syncthreads();

    // prefetch next tile of the weights stream (the HBM-bound traffic)
    if (m0 + BM < mbeg + MSPLIT) {
      const float* wpf = Wm + (size_t)(q0 + lm) * Mn + (m0 + BM) + half * 32;
      __builtin_prefetch(wpf, 0, 1);
    }

    // ---- GEMM1: S = Q * K^T, then t = S * scale * W ----
    v8f t[NT1];
    #pragma unroll
    for (int nt = 0; nt < NT1; ++nt) {
      v8f s = zero;
      #pragma unroll
      for (int kk = 0; kk < NKC; ++kk) {
        // B layout (32x16): lane holds column (memory row) lane&15;
        // K = kk*32 + half*16 + {0..15} -> 16 contiguous f16 from LDS
        const _Float16* kp = &Kt[nt * 16 + lm][0] + kk * 32 + half * 16;
        v16h b;
        #pragma unroll
        for (int j = 0; j < 16; ++j) b[j] = kp[j];
        s = wmma_f16(qa[kk], b, s);
      }
      // elementwise weights (the 134MB stream), coalesced 64B per half-wave
      v8f tv;
      #pragma unroll
      for (int r = 0; r < 8; ++r) {
        float w = Wm[(size_t)(q0 + r + 8 * half) * Mn + (m0 + nt * 16 + lm)];
        tv[r] = s[r] * scale * w;
      }
      t[nt] = tv;
    }

    // ---- online softmax update ----
    float mnew[8], cf[8];
    #pragma unroll
    for (int r = 0; r < 8; ++r) {
      float v = t[0][r];
      #pragma unroll
      for (int nt = 1; nt < NT1; ++nt) v = fmaxf(v, t[nt][r]);
      v = fmaxf(v, __shfl_xor(v, 1, 32));
      v = fmaxf(v, __shfl_xor(v, 2, 32));
      v = fmaxf(v, __shfl_xor(v, 4, 32));
      v = fmaxf(v, __shfl_xor(v, 8, 32));   // reduce within 16-lane half
      mnew[r] = fmaxf(mrow[r], v);
      cf[r]   = __expf(mrow[r] - mnew[r]);
      mrow[r] = mnew[r];
    }

    // P = exp(t - mnew): accumulate row sums, stash P (f16) for layout swap
    #pragma unroll
    for (int nt = 0; nt < NT1; ++nt) {
      #pragma unroll
      for (int r = 0; r < 8; ++r) {
        float p = __expf(t[nt][r] - mnew[r]);
        t[nt][r] = p;
        Plds[wave][r + 8 * half][nt * 16 + lm] = (_Float16)p;
      }
    }
    #pragma unroll
    for (int r = 0; r < 8; ++r) {
      float v = t[0][r];
      #pragma unroll
      for (int nt = 1; nt < NT1; ++nt) v += t[nt][r];
      v += __shfl_xor(v, 1, 32);
      v += __shfl_xor(v, 2, 32);
      v += __shfl_xor(v, 4, 32);
      v += __shfl_xor(v, 8, 32);
      lrow[r] = lrow[r] * cf[r] + v;
    }

    // rescale running O accumulators
    #pragma unroll
    for (int dt = 0; dt < NDT; ++dt) {
      #pragma unroll
      for (int r = 0; r < 8; ++r) o[dt][r] *= cf[r];
    }

    // ---- GEMM2: O += P * V ----
    // P as A-fragments from per-wave LDS (same wave wrote it; DScnt orders it)
    v16h pa[NKC2];
    #pragma unroll
    for (int kc = 0; kc < NKC2; ++kc) {
      const _Float16* pp = &Plds[wave][lm][0] + kc * 32 + half * 16;
      #pragma unroll
      for (int j = 0; j < 8; ++j) {
        pa[kc][j]     = pp[j];
        pa[kc][8 + j] = pp[8 + j];
      }
    }
    #pragma unroll
    for (int dt = 0; dt < NDT; ++dt) {
      #pragma unroll
      for (int kc = 0; kc < NKC2; ++kc) {
        // B layout: lane holds output dim column dt*16 + (lane&15);
        // K (memory rows) = kc*32 + half*16 + {0..15}: contiguous in Vt
        const _Float16* vp = &Vt[dt * 16 + lm][0] + kc * 32 + half * 16;
        v16h b;
        #pragma unroll
        for (int j = 0; j < 16; ++j) b[j] = vp[j];
        o[dt] = wmma_f16(pa[kc], b, o[dt]);
      }
    }
  }

  // ---- write split partials (unnormalized) + row stats ----
  #pragma unroll
  for (int dt = 0; dt < NDT; ++dt) {
    #pragma unroll
    for (int r = 0; r < 8; ++r) {
      size_t row = (size_t)(q0 + r + 8 * half);
      Opart[((size_t)split * Qn + row) * Dn + dt * 16 + lm] = o[dt][r];
    }
  }
  if (lm == 0) {
    #pragma unroll
    for (int r = 0; r < 8; ++r) {
      size_t row = (size_t)(q0 + r + 8 * half);
      Mst[(size_t)split * Qn + row] = mrow[r];
      Lst[(size_t)split * Qn + row] = lrow[r];
    }
  }
}

// Pass 2: merge the SPLITS partials with log-sum-exp weighting.
__global__ __launch_bounds__(256)
void attn_combine(const float* __restrict__ Opart, const float* __restrict__ Mst,
                  const float* __restrict__ Lst, float* __restrict__ out)
{
  int idx = blockIdx.x * 256 + threadIdx.x;   // over Qn*Dn
  int row = idx >> 7;                          // Dn == 128
  float gm = -3.0e30f;
  #pragma unroll
  for (int s = 0; s < SPLITS; ++s) gm = fmaxf(gm, Mst[s * Qn + row]);
  float L = 0.0f, acc = 0.0f;
  #pragma unroll
  for (int s = 0; s < SPLITS; ++s) {
    float c = __expf(Mst[s * Qn + row] - gm);
    L   += Lst[s * Qn + row] * c;
    acc += Opart[(size_t)s * Qn * Dn + idx] * c;
  }
  out[idx] = acc / L;
}

extern "C" void kernel_launch(void* const* d_in, const int* in_sizes, int n_in,
                              void* d_out, int out_size, void* d_ws, size_t ws_size,
                              hipStream_t stream) {
  const float* Qm = (const float*)d_in[0];   // query [4096,128]
  const float* Km = (const float*)d_in[1];   // memory_states [8192,128]
  const float* Wm = (const float*)d_in[2];   // weights [4096,8192]
  float* out = (float*)d_out;                // [4096,128]

  float* Opart = (float*)d_ws;                         // SPLITS*Qn*Dn floats (16 MB)
  float* Mst   = Opart + (size_t)SPLITS * Qn * Dn;     // SPLITS*Qn
  float* Lst   = Mst   + (size_t)SPLITS * Qn;          // SPLITS*Qn

  dim3 grid(Qn / BQ, SPLITS);                // 32 x 8 = 256 workgroups
  attn_partial<<<grid, 256, 0, stream>>>(Qm, Km, Wm, Opart, Mst, Lst);
  attn_combine<<<(Qn * Dn) / 256, 256, 0, stream>>>(Opart, Mst, Lst, out);
}
